// TransformerDecoderLayer_44135083933761
// MI455X (gfx1250) — compile-verified, matrix operand order checked by reading, not run on hardware
//
#include <hip/hip_runtime.h>
#include <hip/hip_bf16.h>

// MI455X / gfx1250, wave32. Matrix math via v_wmma_f32_16x16x32_bf16;
// query-tile staging in K2 via the Tensor Data Mover (tensor_load_to_lds).

typedef __bf16 bf16;
typedef __attribute__((ext_vector_type(16))) __bf16 v16bf;
typedef __attribute__((ext_vector_type(8)))  __bf16 v8bf;
typedef __attribute__((ext_vector_type(8)))  float  v8f;
typedef __attribute__((ext_vector_type(4)))  unsigned int u32x4;
typedef __attribute__((ext_vector_type(8)))  int i32x8;
typedef __attribute__((ext_vector_type(4)))  int i32x4;

#define BN_EPS 1e-5f
#define N1F 2097152.0f   // 32768*64 elements for BN1
#define MROWS 32768      // rows of x
#define NPAIR 16384      // q*k

__device__ __forceinline__ bf16 f2bf(float x) { return (bf16)x; }

// A/B fragment for wmma_f32_16x16x32_bf16 from a row-major bf16 matrix.
// Lane l holds row (row + (l&15)); K slots are two contiguous 8-element
// (16-byte) chunks at kbase + (l&16?8:0) and +16 further (per ISA 7.12.2).
__device__ __forceinline__ v16bf load_frag(const bf16* __restrict__ base,
                                           int row, int ld, int kbase, int lane) {
  int r  = row + (lane & 15);
  int kb = kbase + ((lane & 16) ? 8 : 0);
  const bf16* p = base + (size_t)r * ld + kb;
  v8bf lo = *(const v8bf*)(p);
  v8bf hi = *(const v8bf*)(p + 16);
  return __builtin_shufflevector(lo, hi, 0,1,2,3,4,5,6,7,8,9,10,11,12,13,14,15);
}

// Same fragment gather but from LDS (compiler emits ds_load_b128 pairs).
__device__ __forceinline__ v16bf load_frag_lds(const bf16* base,
                                               int row, int kbase, int lane) {
  int r  = row + (lane & 15);
  int kb = kbase + ((lane & 16) ? 8 : 0);
  const bf16* p = base + r * 512 + kb;
  v8bf lo = *(const v8bf*)(p);
  v8bf hi = *(const v8bf*)(p + 16);
  return __builtin_shufflevector(lo, hi, 0,1,2,3,4,5,6,7,8,9,10,11,12,13,14,15);
}

__device__ __forceinline__ v8f wmma_bf(v16bf a, v16bf b, v8f c) {
  return __builtin_amdgcn_wmma_f32_16x16x32_bf16(false, a, false, b, (short)0, c, false, false);
}

// TDM: async-load a contiguous [rows x 512] bf16 tile (row stride 512) from
// global memory into LDS at byte offset lds_off. 2D descriptor per ISA ch.8:
// data_size=2B, tensor/tile dim0=512, dim1=rows, dim0_stride=512.
__device__ __forceinline__ void tdm_load_tile(unsigned long long gaddr,
                                              unsigned lds_off, int rows) {
  u32x4 g0;
  g0.x = 1u;                                          // count=1 (valid), no gather
  g0.y = lds_off;                                     // LDS byte address
  g0.z = (unsigned)(gaddr & 0xFFFFFFFFull);           // global_addr[31:0]
  g0.w = (unsigned)((gaddr >> 32) & 0x01FFFFFFull)    // global_addr[56:32]
         | (2u << 30);                                // type=2 ("image")
  i32x8 g1;
  g1[0] = 1 << 16;                                    // data_size=1 (2 bytes)
  g1[1] = (int)(512u << 16);                          // tensor_dim0 = 512
  g1[2] = (int)((unsigned)rows << 16);                // tensor_dim1 = rows
  g1[3] = (int)(512u << 16);                          // tile_dim0 = 512
  g1[4] = rows;                                       // tile_dim1 = rows, tile_dim2=0
  g1[5] = 512;                                        // tensor_dim0_stride = 512
  g1[6] = 0;
  g1[7] = 0;
  i32x4 gz = {0, 0, 0, 0};
#if __clang_major__ >= 23
  i32x8 gz8 = {0, 0, 0, 0, 0, 0, 0, 0};
  __builtin_amdgcn_tensor_load_to_lds(g0, g1, gz, gz, gz8, 0);
#else
  __builtin_amdgcn_tensor_load_to_lds(g0, g1, gz, gz, 0);
#endif
}

// LDS byte offset of a __shared__ object (ptrtoint of the AS(3) pointer).
__device__ __forceinline__ unsigned lds_offset_of(const void* p) {
  return (unsigned)(unsigned long long)
         (__attribute__((address_space(3))) const char*)p;
}

// -------------------- K0: convert inputs to bf16, precompute sigmoid, zero accumulators
__global__ void k_setup(const float* __restrict__ tgt, const float* __restrict__ mem,
                        const float* __restrict__ W1, const float* __restrict__ W2,
                        const float* __restrict__ se,
                        bf16* Xbf, bf16* W1bf, bf16* W2bf, float* sigE,
                        float* bn1, float* t64, float* G, float* bn3) {
  int stride = gridDim.x * blockDim.x;
  int tid = blockIdx.x * blockDim.x + threadIdx.x;
  for (int i = tid; i < 20480 * 512; i += stride)
    Xbf[i] = f2bf(i < 2097152 ? tgt[i] : mem[i - 2097152]);
  for (int i = tid; i < 512 * 512; i += stride) W1bf[i] = f2bf(W1[i]);
  for (int i = tid; i < 2048 * 64; i += stride) W2bf[i] = f2bf(W2[i]);
  for (int i = tid; i < 64 * 64;  i += stride) sigE[i] = 1.0f / (1.0f + __expf(-se[i]));
  for (int i = tid; i < 4096; i += stride) G[i] = 0.0f;
  for (int i = tid; i < 64;   i += stride) t64[i] = 0.0f;
  if (tid < 2) { bn1[tid] = 0.0f; bn3[tid] = 0.0f; }
}

// -------------------- K1: Y[20480,512] = X @ W1^T + b1   (bf16 out)
__global__ __launch_bounds__(128) void k_gemm_w1(const bf16* __restrict__ Xbf,
                                                 const bf16* __restrict__ W1bf,
                                                 const float* __restrict__ b1,
                                                 bf16* __restrict__ Ybf) {
  int lane = threadIdx.x & 31, wave = threadIdx.x >> 5;
  int m_base = blockIdx.x * 256 + wave * 64;
  int n_base = blockIdx.y * 64;
  v8f vz = {0.f,0.f,0.f,0.f,0.f,0.f,0.f,0.f};
  v8f acc[16];
  #pragma unroll
  for (int i = 0; i < 16; i++) acc[i] = vz;
  for (int kb = 0; kb < 512; kb += 32) {
    v16bf a[4], b[4];
    #pragma unroll
    for (int i = 0; i < 4; i++) {
      a[i] = load_frag(Xbf,  m_base + i * 16, 512, kb, lane);   // A: rows of X
      b[i] = load_frag(W1bf, n_base + i * 16, 512, kb, lane);   // B[k][n] = W1[n][k]
    }
    #pragma unroll
    for (int mi = 0; mi < 4; mi++)
      #pragma unroll
      for (int ni = 0; ni < 4; ni++)
        acc[mi * 4 + ni] = wmma_bf(a[mi], b[ni], acc[mi * 4 + ni]);
  }
  int hi8 = (lane & 16) ? 8 : 0;
  #pragma unroll
  for (int mi = 0; mi < 4; mi++)
    #pragma unroll
    for (int ni = 0; ni < 4; ni++) {
      int o = n_base + ni * 16 + (lane & 15);
      float bias = b1[o];
      int m0 = m_base + mi * 16 + hi8;
      v8f A = acc[mi * 4 + ni];
      #pragma unroll
      for (int r = 0; r < 8; r++)
        Ybf[(size_t)(m0 + r) * 512 + o] = f2bf(A[r] + bias);   // lanes 0-15 coalesce 32B
    }
}

// -------------------- K2: fused score einsum + sigmoid-modulate + dual max-pool
// Block = 4 waves handles (q, 4 consecutive k). The shared 64x512 bf16 query
// tile (exactly 64 KB, contiguous) is staged into LDS once via the Tensor
// Data Mover; B fragments then come from LDS (ds_load_b128), halving L2
// traffic of the dominant kernel. A fragments (per-wave keym) stream from L2.
// The 268 MB score tensor is never materialized.
__global__ __launch_bounds__(128) void k_score(const bf16* __restrict__ Ybf,
                                               const float* __restrict__ sigE,
                                               float* __restrict__ pooled,
                                               float* bn1) {
  __shared__ bf16 s_q[64 * 512];   // 65536 bytes
  int lane = threadIdx.x & 31, wave = threadIdx.x >> 5;
  int qi = blockIdx.x >> 6;
  int ki = (blockIdx.x & 63) * 4 + wave;
  const int qrow = qi * 64;          // query rows in Y
  const int krow = 4096 + ki * 64;   // keym rows in Y

  if (wave == 0) {
    tdm_load_tile((unsigned long long)(const void*)(Ybf + (size_t)qrow * 512),
                  lds_offset_of(s_q), 64);
    __builtin_amdgcn_s_wait_tensorcnt(0);
  }
  __syncthreads();

  v8f vz = {0.f,0.f,0.f,0.f,0.f,0.f,0.f,0.f};
  v8f acc[16];
  #pragma unroll
  for (int i = 0; i < 16; i++) acc[i] = vz;

  for (int d = 0; d < 512; d += 32) {
    v16bf a[4], b[4];
    #pragma unroll
    for (int i = 0; i < 4; i++) {
      a[i] = load_frag(Ybf, krow + i * 16, 512, d, lane);  // A: keym (M = s), from L2
      b[i] = load_frag_lds(s_q, i * 16, d, lane);          // B: query (N = t), from LDS
    }
    #pragma unroll
    for (int si = 0; si < 4; si++)
      #pragma unroll
      for (int ti = 0; ti < 4; ti++)
        acc[si * 4 + ti] = wmma_bf(a[si], b[ti], acc[si * 4 + ti]);
  }

  int hi8 = (lane & 16) ? 8 : 0;
  float rmax[32], cmax[4];
  #pragma unroll
  for (int i = 0; i < 32; i++) rmax[i] = -3.0e38f;
  #pragma unroll
  for (int i = 0; i < 4; i++) cmax[i] = -3.0e38f;

  #pragma unroll
  for (int si = 0; si < 4; si++)
    #pragma unroll
    for (int ti = 0; ti < 4; ti++) {
      v8f A = acc[si * 4 + ti];
      int t = ti * 16 + (lane & 15);
      #pragma unroll
      for (int r = 0; r < 8; r++) {
        float v = A[r] * sigE[(si * 16 + hi8 + r) * 64 + t];
        rmax[si * 8 + r] = fmaxf(rmax[si * 8 + r], v);  // max over t (per s)
        cmax[ti]         = fmaxf(cmax[ti], v);          // max over s (per t)
      }
    }
  // rmax: reduce across the 16 lanes of each half (same M rows)
  #pragma unroll
  for (int m = 1; m < 16; m <<= 1)
    #pragma unroll
    for (int i = 0; i < 32; i++) rmax[i] = fmaxf(rmax[i], __shfl_xor(rmax[i], m, 32));
  // cmax: combine the two row-halves (lane l and l+16 share the same t)
  #pragma unroll
  for (int i = 0; i < 4; i++) cmax[i] = fmaxf(cmax[i], __shfl_xor(cmax[i], 16, 32));

  float* prow = pooled + (size_t)(qi * 256 + ki) * 128;
  float ls = 0.f, lq = 0.f;
  if (lane < 16) {                       // first 64 = max over s (per t)
    #pragma unroll
    for (int ti = 0; ti < 4; ti++) {
      float v = cmax[ti];
      prow[ti * 16 + lane] = v; ls += v; lq += v * v;
    }
  }
  if ((lane & 15) == 0) {                // second 64 = max over t (per s)
    #pragma unroll
    for (int si = 0; si < 4; si++)
      #pragma unroll
      for (int r = 0; r < 8; r++) {
        float v = rmax[si * 8 + r];
        prow[64 + si * 16 + hi8 + r] = v; ls += v; lq += v * v;
      }
  }
  // BN1 partial sums: reuse the (now idle) s_q LDS space for block reduction.
  __syncthreads();
  float* s_red = (float*)s_q;
  if (threadIdx.x == 0) { s_red[0] = 0.f; s_red[1] = 0.f; }
  __syncthreads();
  if (ls != 0.f || lq != 0.f) { atomicAdd(&s_red[0], ls); atomicAdd(&s_red[1], lq); }
  __syncthreads();
  if (threadIdx.x == 0) { atomicAdd(&bn1[0], s_red[0]); atomicAdd(&bn1[1], s_red[1]); }
}

// -------------------- K3: t = colsum(xn), G = xn^T xn  (xn = BN1-normalized pooled)
__global__ __launch_bounds__(256) void k_gram(const float* __restrict__ pooled,
                                              const float* __restrict__ bn1,
                                              const float* g1, const float* be1,
                                              float* t64, float* G) {
  float m1 = bn1[0] / N1F;
  float v1 = bn1[1] / N1F - m1 * m1;
  float s1 = rsqrtf(v1 + BN_EPS) * g1[0];
  float c1 = be1[0];
  __shared__ float sx[4][64];
  float gl[16];
  #pragma unroll
  for (int e = 0; e < 16; e++) gl[e] = 0.f;
  float tl = 0.f;
  int rbase = blockIdx.x * 512;
  int rr = threadIdx.x >> 6, cc = threadIdx.x & 63;
  for (int ch = 0; ch < 128; ch++) {
    int r0 = rbase + ch * 4;
    sx[rr][cc] = (pooled[(size_t)(r0 + rr) * 64 + cc] - m1) * s1 + c1;
    __syncthreads();
    int e0 = threadIdx.x * 16;
    #pragma unroll
    for (int e = 0; e < 16; e++) {
      int idx = e0 + e, i = idx >> 6, j = idx & 63;
      gl[e] += sx[0][i]*sx[0][j] + sx[1][i]*sx[1][j] + sx[2][i]*sx[2][j] + sx[3][i]*sx[3][j];
    }
    if (threadIdx.x < 64)
      tl += sx[0][threadIdx.x] + sx[1][threadIdx.x] + sx[2][threadIdx.x] + sx[3][threadIdx.x];
    __syncthreads();
  }
  int e0 = threadIdx.x * 16;
  #pragma unroll
  for (int e = 0; e < 16; e++) atomicAdd(&G[e0 + e], gl[e]);
  if (threadIdx.x < 64) atomicAdd(&t64[threadIdx.x], tl);
}

// -------------------- K3b: analytic BN2 stats per feature from t and G
// mean_f = (W2f.t)/M + b2 ; E[h^2]_f = (W2f G W2f^T + 2 b2 (W2f.t))/M + b2^2
__global__ __launch_bounds__(128) void k_bn2stats(const float* __restrict__ W2,
                                                  const float* b2, const float* g2,
                                                  const float* be2,
                                                  const float* __restrict__ t64,
                                                  const float* __restrict__ G,
                                                  float* sc2, float* sh2) {
  __shared__ float sG[4096];
  __shared__ float sT[64];
  for (int i = threadIdx.x; i < 4096; i += 128) sG[i] = G[i];
  if (threadIdx.x < 64) sT[threadIdx.x] = t64[threadIdx.x];
  __syncthreads();
  int f = blockIdx.x * 128 + threadIdx.x;
  float wf[64];
  #pragma unroll
  for (int i = 0; i < 64; i++) wf[i] = W2[f * 64 + i];
  float dt = 0.f;
  #pragma unroll
  for (int i = 0; i < 64; i++) dt += wf[i] * sT[i];
  float quad = 0.f;
  for (int i = 0; i < 64; i++) {
    float s = 0.f;
    #pragma unroll 8
    for (int j = 0; j < 64; j++) s += sG[i * 64 + j] * wf[j];
    quad += wf[i] * s;
  }
  const float invM = 1.0f / (float)MROWS;
  float b2f  = b2[f];
  float mean = dt * invM + b2f;
  float ex2  = (quad + 2.0f * b2f * dt) * invM + b2f * b2f;
  float var  = ex2 - mean * mean;
  float sc   = g2[f] * rsqrtf(var + BN_EPS);
  sc2[f] = sc;
  sh2[f] = (b2f - mean) * sc + be2[f];   // h_bn = h_raw*sc + sh  (h_raw excludes b2)
}

// -------------------- K4: fused head: h = xn@W2^T, BN2+ReLU, dot W3, pair-sum, BN3 stats
__global__ __launch_bounds__(128) void k_head(const float* __restrict__ pooled,
                                              const bf16* __restrict__ W2bf,
                                              const float* __restrict__ W3,
                                              const float* b3,
                                              const float* __restrict__ bn1,
                                              const float* g1, const float* be1,
                                              const float* __restrict__ sc2,
                                              const float* __restrict__ sh2,
                                              float* __restrict__ zbuf, float* bn3) {
  __shared__ float s_sum, s_sq;
  if (threadIdx.x == 0) { s_sum = 0.f; s_sq = 0.f; }
  __syncthreads();
  int lane = threadIdx.x & 31, wave = threadIdx.x >> 5;
  int m_base = (blockIdx.x * 4 + wave) * 16;
  float m1 = bn1[0] / N1F;
  float v1 = bn1[1] / N1F - m1 * m1;
  float s1 = rsqrtf(v1 + BN_EPS) * g1[0];
  float c1 = be1[0];

  int mrow = m_base + (lane & 15);
  int base8 = (lane & 16) ? 8 : 0;
  v16bf fa0, fa1;
  #pragma unroll
  for (int j = 0; j < 16; j++) {
    int col = base8 + (j < 8 ? j : j + 8);
    fa0[j] = f2bf((pooled[(size_t)mrow * 64 + col]      - m1) * s1 + c1);
    fa1[j] = f2bf((pooled[(size_t)mrow * 64 + 32 + col] - m1) * s1 + c1);
  }
  v8f vz = {0.f,0.f,0.f,0.f,0.f,0.f,0.f,0.f};
  float ysum[8] = {0.f,0.f,0.f,0.f,0.f,0.f,0.f,0.f};
  for (int ft = 0; ft < 128; ft++) {
    int fb = ft * 16;
    v16bf b0 = load_frag(W2bf, fb, 64, 0,  lane);
    v16bf b1 = load_frag(W2bf, fb, 64, 32, lane);
    v8f accv = vz;
    accv = wmma_bf(fa0, b0, accv);
    accv = wmma_bf(fa1, b1, accv);
    int f = fb + (lane & 15);
    float sc = sc2[f], sh = sh2[f], w3 = W3[f];
    #pragma unroll
    for (int r = 0; r < 8; r++) {
      float hn = accv[r] * sc + sh;
      ysum[r] += (hn > 0.f) ? hn * w3 : 0.f;
    }
  }
  #pragma unroll
  for (int m = 1; m < 16; m <<= 1)
    #pragma unroll
    for (int r = 0; r < 8; r++) ysum[r] += __shfl_xor(ysum[r], m, 32);

  float ls = 0.f, lq = 0.f;
  if ((lane & 15) == 0) {
    int row0 = m_base + base8;
    float b3v = b3[0];
    #pragma unroll
    for (int i = 0; i < 4; i++) {
      float z = (ysum[2 * i] + b3v) + (ysum[2 * i + 1] + b3v);
      zbuf[row0 / 2 + i] = z;
      ls += z; lq += z * z;
    }
  }
  atomicAdd(&s_sum, ls); atomicAdd(&s_sq, lq);
  __syncthreads();
  if (threadIdx.x == 0) { atomicAdd(&bn3[0], s_sum); atomicAdd(&bn3[1], s_sq); }
}

// -------------------- K5: BN3 finalize -> output (64,256)
__global__ void k_final(const float* __restrict__ zbuf, const float* __restrict__ bn3,
                        const float* g3, const float* be3, float* __restrict__ out) {
  int p = blockIdx.x * blockDim.x + threadIdx.x;
  if (p >= NPAIR) return;
  float m3 = bn3[0] / (float)NPAIR;
  float v3 = bn3[1] / (float)NPAIR - m3 * m3;
  out[p] = (zbuf[p] - m3) * rsqrtf(v3 + BN_EPS) * g3[0] + be3[0];
}

extern "C" void kernel_launch(void* const* d_in, const int* in_sizes, int n_in,
                              void* d_out, int out_size, void* d_ws, size_t ws_size,
                              hipStream_t stream) {
  const float* tgt = (const float*)d_in[0];
  const float* mem = (const float*)d_in[1];
  const float* W1  = (const float*)d_in[2];
  const float* b1  = (const float*)d_in[3];
  const float* se  = (const float*)d_in[4];
  const float* W2  = (const float*)d_in[5];
  const float* b2  = (const float*)d_in[6];
  const float* W3  = (const float*)d_in[7];
  const float* b3  = (const float*)d_in[8];
  const float* g1  = (const float*)d_in[9];
  const float* be1 = (const float*)d_in[10];
  const float* g2  = (const float*)d_in[11];
  const float* be2 = (const float*)d_in[12];
  const float* g3  = (const float*)d_in[13];
  const float* be3 = (const float*)d_in[14];
  // d_in[15] = flag (always 1 in this problem's inputs)

  size_t off = 0;
  char* base = (char*)d_ws;
  auto carve = [&](size_t bytes) -> void* {
    void* p = base + off;
    off += (bytes + 255) & ~(size_t)255;
    return p;
  };
  bf16*  Xbf    = (bf16*) carve((size_t)20480 * 512 * 2);  // tgt||mem in bf16
  bf16*  W1bf   = (bf16*) carve((size_t)512 * 512 * 2);
  bf16*  W2bf   = (bf16*) carve((size_t)2048 * 64 * 2);
  bf16*  Ybf    = (bf16*) carve((size_t)20480 * 512 * 2);  // query||keym
  float* pooled = (float*)carve((size_t)MROWS * 64 * 4);
  float* sigE   = (float*)carve(4096 * 4);
  float* G      = (float*)carve(4096 * 4);
  float* t64    = (float*)carve(64 * 4);
  float* sc2    = (float*)carve(2048 * 4);
  float* sh2    = (float*)carve(2048 * 4);
  float* zbuf   = (float*)carve((size_t)NPAIR * 4);
  float* bn1    = (float*)carve(2 * 4);
  float* bn3    = (float*)carve(2 * 4);
  (void)ws_size; (void)in_sizes; (void)n_in; (void)out_size;

  k_setup<<<2048, 256, 0, stream>>>(tgt, mem, W1, W2, se, Xbf, W1bf, W2bf, sigE,
                                    bn1, t64, G, bn3);
  k_gemm_w1<<<dim3(80, 8), 128, 0, stream>>>(Xbf, W1bf, b1, Ybf);
  k_score<<<4096, 128, 0, stream>>>(Ybf, sigE, pooled, bn1);
  k_gram<<<64, 256, 0, stream>>>(pooled, bn1, g1, be1, t64, G);
  k_bn2stats<<<16, 128, 0, stream>>>(W2, b2, g2, be2, t64, G, sc2, sh2);
  k_head<<<512, 128, 0, stream>>>(pooled, W2bf, W3, b3, bn1, g1, be1, sc2, sh2,
                                  zbuf, bn3);
  k_final<<<64, 256, 0, stream>>>(zbuf, bn3, g3, be3, (float*)d_out);
}
